// EMAVectorQuantizer_6141803233292
// MI455X (gfx1250) — matrix operand order repaired and checked by reading, not compiled, not run on hardware
//
#include <hip/hip_runtime.h>
#include <hip/hip_bf16.h>

// ---------------------------------------------------------------------------
// EMA Vector Quantizer for MI455X (gfx1250, wave32, WMMA + async-LDS)
//
//   z:          [16, 256, 32, 32] f32   -> NVEC = 16384 tokens of dim 256
//   embedding:  [8192, 256] f32         -> K = 8192 codes
//   outputs: out(4194304) | loss(1) | idx(16384) | new_cluster_size(8192)
//            | new_ema_w(2097152) | new_embedding(2097152)
// ---------------------------------------------------------------------------

#define NVEC      16384
#define KCODES    8192
#define DIM       256
#define TILE_N    32         // codes per staged LDS tile
#define BETA      0.25f
#define DECAY     0.99f
#define ONE_M_D   0.01f
#define EPSV      1e-05f

typedef __attribute__((ext_vector_type(16))) __bf16 v16bf;
typedef __attribute__((ext_vector_type(8)))  __bf16 v8bf;
typedef __attribute__((ext_vector_type(8)))  float  v8f;
typedef __attribute__((ext_vector_type(4)))  int    v4i;

union ABfrag { v16bf v; v8bf h[2]; };

typedef unsigned short ushort_t;

#if defined(__gfx1250__) && __has_builtin(__builtin_amdgcn_global_load_async_to_lds_b128)
#define HAVE_ASYNC_LDS 1
#else
#define HAVE_ASYNC_LDS 0
#endif

// output section offsets (floats)
#define OUT_OFF   0
#define LOSS_OFF  4194304
#define IDX_OFF   4194305
#define NCS_OFF   (4194305 + 16384)              // 4210689
#define EMA_OFF   (NCS_OFF + 8192)               // 4218881
#define EMB_OFF   (EMA_OFF + 2097152)            // 6316033

// workspace layout (bytes)
#define WS_EBF     0u                            // K*256 bf16  = 4 MiB
#define WS_ZBF     (4u*1024u*1024u)              // N*256 bf16  = 8 MiB
#define WS_ENORM   (12u*1024u*1024u)             // K f32  = 32 KiB
#define WS_COUNTS  (WS_ENORM + 32u*1024u)        // K f32  = 32 KiB
#define WS_CS      (WS_COUNTS + 32u*1024u)       // K f32  = 32 KiB
#define WS_IDX     (WS_CS + 32u*1024u)           // N i32  = 64 KiB
#define WS_SCAL    (WS_IDX + 64u*1024u)          // [0]=loss accum, [1]=cs sum

__device__ __forceinline__ ushort_t f2bf(float f) {
  unsigned int u = __float_as_uint(f);
  unsigned int r = u + 0x7FFFu + ((u >> 16) & 1u);   // round-to-nearest-even
  return (ushort_t)(r >> 16);
}

// ---------------------------------------------------------------------------
// Kernel 1: embedding -> bf16 copy + row squared norms. One block per row.
// ---------------------------------------------------------------------------
__global__ void k_prep_e(const float* __restrict__ emb,
                         ushort_t* __restrict__ ebf,
                         float* __restrict__ enorm) {
  __shared__ float s[256];
  int row = blockIdx.x;
  int d = threadIdx.x;
  float v = emb[row * DIM + d];
  ebf[row * DIM + d] = f2bf(v);
  s[d] = v * v;
  __syncthreads();
#pragma unroll
  for (int off = 128; off > 0; off >>= 1) {
    if (d < off) s[d] += s[d + off];
    __syncthreads();
  }
  if (d == 0) enorm[row] = s[0];
}

// ---------------------------------------------------------------------------
// Kernel 2: z [B,D,H,W] -> zbf [N,256] bf16 (transpose). Lane -> token so
// global reads are coalesced. (No norm needed: ||z||^2 is argmin-invariant.)
// ---------------------------------------------------------------------------
__global__ void k_prep_z(const float* __restrict__ z,
                         ushort_t* __restrict__ zbf) {
  int l = threadIdx.x & 31;
  int w = threadIdx.x >> 5;
  int n = blockIdx.x * 32 + l;
  int nb = n >> 10;           // batch
  int nr = n & 1023;          // h*32+w
  const float* zbase = z + (size_t)nb * (DIM * 1024) + nr;
  for (int d = w; d < DIM; d += 8) {
    zbf[(size_t)n * DIM + d] = f2bf(zbase[(size_t)d * 1024]);
  }
}

// ---------------------------------------------------------------------------
// Kernel 3: seed new_ema_w output with DECAY * ema_w (scatter adds 0.01*z).
// ---------------------------------------------------------------------------
__global__ void k_init_ema(const float* __restrict__ ema_w,
                           float* __restrict__ ema_out) {
  int i = blockIdx.x * 256 + threadIdx.x;
  ema_out[i] = ema_w[i] * DECAY;
}

// ---------------------------------------------------------------------------
// Kernel 4: fused GEMM + argmin.
//   256 threads = 8 waves; wave owns a register-resident 16x256 bf16 A panel.
//   256 iterations over 32-code E tiles staged in LDS (async + double buffer
//   when available), shared by all 8 waves. Per iteration: 16 chained
//   v_wmma_f32_16x16x32_bf16, then per-lane running argmin of
//   key = ||e||^2 - 2*(z.e)  (|z|^2 dropped: constant per token).
// ---------------------------------------------------------------------------
__global__ void __launch_bounds__(256)
k_vq_main(const ushort_t* __restrict__ zbf,
          const ushort_t* __restrict__ ebf,
          const float* __restrict__ enorm,
          int* __restrict__ idx_out,
          float* __restrict__ idx_out_f,
          float* __restrict__ counts) {
  __shared__ __align__(16) ushort_t ldsB[2][TILE_N * DIM];   // 2 x 16 KiB

  const int tid  = threadIdx.x;
  const int wave = tid >> 5;
  const int lane = tid & 31;
  const int m    = lane & 15;       // token row inside A tile
  const int half = lane >> 4;       // lane half selects K sub-range
  const int n    = lane & 15;       // output column inside C tile
  const int t0   = blockIdx.x * 128 + wave * 16;

  // --- A panel: 16 tokens x 256 dims, register resident ---------------------
  // A 16x32 bf16 layout: lane holds row m; K chunks [h*8..h*8+7], [h*8+16..+23].
  ABfrag a[8];
  {
    const ushort_t* arow = zbf + (size_t)(t0 + m) * DIM;
#pragma unroll
    for (int kc = 0; kc < 8; ++kc) {
      const ushort_t* p = arow + kc * 32 + half * 8;
      a[kc].h[0] = *(const v8bf*)(p);
      a[kc].h[1] = *(const v8bf*)(p + 16);
    }
  }

  float minv[8];
  int   mini[8];
#pragma unroll
  for (int r = 0; r < 8; ++r) { minv[r] = 3.4e38f; mini[r] = 0; }

  // cooperative copy mapping: thread -> (row, 64B chunk) of the 16 KiB tile
  const int ldrow = tid >> 3;          // 0..31
  const int ldcol = (tid & 7) * 32;    // ushort offset within row

  // one 16x16 WMMA pair per kc-chunk, two code sub-tiles per iteration
  auto compute_tile = [&](const ushort_t* buf, int j) {
    float en0 = enorm[j + n];
    float en1 = enorm[j + 16 + n];
    v8f c0 = {};
    v8f c1 = {};
#pragma unroll
    for (int kc = 0; kc < 8; ++kc) {
      ABfrag b0, b1;
      // B 32x16 bf16 layout: lane holds column n; K chunks [h*8..h*8+7], [h*8+16..+23]
      const ushort_t* p = buf + n * DIM + kc * 32 + half * 8;
      b0.h[0] = *(const v8bf*)(p);
      b0.h[1] = *(const v8bf*)(p + 16);
      const ushort_t* q = p + 16 * DIM;
      b1.h[0] = *(const v8bf*)(q);
      b1.h[1] = *(const v8bf*)(q + 16);
      c0 = __builtin_amdgcn_wmma_f32_16x16x32_bf16(
               false, a[kc].v, false, b0.v, (short)0, c0, false, false);
      c1 = __builtin_amdgcn_wmma_f32_16x16x32_bf16(
               false, a[kc].v, false, b1.v, (short)0, c1, false, false);
    }
    // C layout: VGPR r -> M = r + 8*half, N = n
#pragma unroll
    for (int r = 0; r < 8; ++r) {
      float k0 = fmaf(-2.0f, c0[r], en0);
      if (k0 < minv[r]) { minv[r] = k0; mini[r] = j + n; }
      float k1 = fmaf(-2.0f, c1[r], en1);
      if (k1 < minv[r]) { minv[r] = k1; mini[r] = j + 16 + n; }
    }
  };

#if HAVE_ASYNC_LDS
  // --- async double-buffered pipeline: 1 barrier / iteration ---------------
  auto stage_async = [&](int j, ushort_t* buf) {
    v4i* g = (v4i*)(ebf + (size_t)(j + ldrow) * DIM + ldcol);
    v4i* l = (v4i*)(buf + ldrow * DIM + ldcol);
    __builtin_amdgcn_global_load_async_to_lds_b128(g,     l,     0, 0);
    __builtin_amdgcn_global_load_async_to_lds_b128(g + 1, l + 1, 0, 0);
    __builtin_amdgcn_global_load_async_to_lds_b128(g + 2, l + 2, 0, 0);
    __builtin_amdgcn_global_load_async_to_lds_b128(g + 3, l + 3, 0, 0);
  };

  ushort_t* bufA = &ldsB[0][0];
  ushort_t* bufB = &ldsB[1][0];
  stage_async(0, bufA);
  for (int j = 0; j < KCODES; j += TILE_N) {
    asm volatile("s_wait_asynccnt 0x0" ::: "memory");  // my tile writes done
    __syncthreads();                                   // all writes done + prev reads done
    if (j + TILE_N < KCODES) stage_async(j + TILE_N, bufB);
    compute_tile(bufA, j);
    ushort_t* t = bufA; bufA = bufB; bufB = t;
  }
#else
  // --- fallback: reg-staged cooperative copy, single buffer ----------------
  for (int j = 0; j < KCODES; j += TILE_N) {
    const v8bf* src = (const v8bf*)(ebf + (size_t)(j + ldrow) * DIM + ldcol);
    v8bf r0 = src[0], r1 = src[1], r2 = src[2], r3 = src[3];
    __syncthreads();                 // previous tile fully consumed
    v8bf* dst = (v8bf*)(&ldsB[0][0] + ldrow * DIM + ldcol);
    dst[0] = r0; dst[1] = r1; dst[2] = r2; dst[3] = r3;
    __syncthreads();                 // tile visible to all waves
    compute_tile(&ldsB[0][0], j);
  }
#endif

  // cross-lane reduction over the 16 N-lanes (xor masks stay in 16-group)
#pragma unroll
  for (int r = 0; r < 8; ++r) {
#pragma unroll
    for (int msk = 8; msk >= 1; msk >>= 1) {
      float ov = __shfl_xor(minv[r], msk, 32);
      int   oi = __shfl_xor(mini[r], msk, 32);
      if (ov < minv[r] || (ov == minv[r] && oi < mini[r])) {
        minv[r] = ov; mini[r] = oi;
      }
    }
  }

  if ((lane & 15) == 0) {
#pragma unroll
    for (int r = 0; r < 8; ++r) {
      int token = t0 + r + 8 * half;
      idx_out[token]   = mini[r];
      idx_out_f[token] = (float)mini[r];
      atomicAdd(&counts[mini[r]], 1.0f);
    }
  }
}

// ---------------------------------------------------------------------------
// Kernel 5: gather z_q = E[idx] -> out (transposed back), loss partials,
//           embed_sum atomics into pre-scaled ema output.
//           i = d*16384 + n so lanes vary n => coalesced z/out traffic.
// ---------------------------------------------------------------------------
__global__ void k_scatter(const float* __restrict__ z,
                          const float* __restrict__ emb,
                          const int* __restrict__ idx,
                          float* __restrict__ out,
                          float* __restrict__ ema_out,
                          float* __restrict__ loss_acc) {
  __shared__ float s[256];
  int i = blockIdx.x * 256 + threadIdx.x;
  int nn = i & (NVEC - 1);
  int d  = i >> 14;
  int k  = idx[nn];
  float ev = emb[(size_t)k * DIM + d];
  size_t zaddr = (size_t)(nn >> 10) * (DIM * 1024) + (size_t)d * 1024 + (nn & 1023);
  float zv = z[zaddr];
  out[zaddr] = ev;                       // straight-through output == z_q
  float diff = ev - zv;
  s[threadIdx.x] = diff * diff;
  atomicAdd(&ema_out[(size_t)k * DIM + d], ONE_M_D * zv);
  __syncthreads();
#pragma unroll
  for (int off = 128; off > 0; off >>= 1) {
    if (threadIdx.x < off) s[threadIdx.x] += s[threadIdx.x + off];
    __syncthreads();
  }
  if (threadIdx.x == 0) atomicAdd(loss_acc, s[0]);
}

// ---------------------------------------------------------------------------
// Kernel 6: cs = decay*cluster_size + 0.01*counts + eps ; global sum.
// ---------------------------------------------------------------------------
__global__ void k_cs(const float* __restrict__ cluster_size,
                     const float* __restrict__ counts,
                     float* __restrict__ cs,
                     float* __restrict__ cs_sum) {
  __shared__ float s[256];
  int k = blockIdx.x * 256 + threadIdx.x;
  float c = cluster_size[k] * DECAY + ONE_M_D * counts[k] + EPSV;
  cs[k] = c;
  s[threadIdx.x] = c;
  __syncthreads();
#pragma unroll
  for (int off = 128; off > 0; off >>= 1) {
    if (threadIdx.x < off) s[threadIdx.x] += s[threadIdx.x + off];
    __syncthreads();
  }
  if (threadIdx.x == 0) atomicAdd(cs_sum, s[0]);
}

// ---------------------------------------------------------------------------
// Kernel 7: normalize cluster sizes, divide ema -> new embedding, loss scale.
// ---------------------------------------------------------------------------
__global__ void k_final(const float* __restrict__ cs,
                        const float* __restrict__ scal,   // [0]=loss, [1]=cs sum
                        const float* __restrict__ ema_out,
                        float* __restrict__ ncs_out,
                        float* __restrict__ emb_out,
                        float* __restrict__ loss_out) {
  int i = blockIdx.x * 256 + threadIdx.x;
  int k = i >> 8;
  int d = i & 255;
  float denom = scal[1] + (float)KCODES * EPSV;
  float ncs = cs[k] / denom;
  if (d == 0) ncs_out[k] = ncs;
  emb_out[i] = ema_out[i] / ncs;
  if (i == 0) loss_out[0] = BETA * scal[0] / (float)((size_t)NVEC * DIM);
}

// ---------------------------------------------------------------------------
extern "C" void kernel_launch(void* const* d_in, const int* in_sizes, int n_in,
                              void* d_out, int out_size, void* d_ws, size_t ws_size,
                              hipStream_t stream) {
  (void)in_sizes; (void)n_in; (void)out_size; (void)ws_size;

  const float* z            = (const float*)d_in[0];
  const float* emb          = (const float*)d_in[1];
  const float* cluster_size = (const float*)d_in[2];
  const float* ema_w        = (const float*)d_in[3];

  float* out = (float*)d_out;
  char*  ws  = (char*)d_ws;

  ushort_t* ebf    = (ushort_t*)(ws + WS_EBF);
  ushort_t* zbf    = (ushort_t*)(ws + WS_ZBF);
  float*    enorm  = (float*)(ws + WS_ENORM);
  float*    counts = (float*)(ws + WS_COUNTS);
  float*    cs     = (float*)(ws + WS_CS);
  int*      idx    = (int*)(ws + WS_IDX);
  float*    scal   = (float*)(ws + WS_SCAL);   // [0]=loss accum, [1]=cs sum

  float* out_main = out + OUT_OFF;
  float* out_loss = out + LOSS_OFF;
  float* out_idx  = out + IDX_OFF;
  float* out_ncs  = out + NCS_OFF;
  float* out_ema  = out + EMA_OFF;
  float* out_emb  = out + EMB_OFF;

  // zero the atomic accumulators (graph-capturable async memsets)
  (void)hipMemsetAsync(counts, 0, KCODES * sizeof(float), stream);
  (void)hipMemsetAsync(scal, 0, 2 * sizeof(float), stream);

  k_prep_e  <<<KCODES, 256, 0, stream>>>(emb, ebf, enorm);
  k_prep_z  <<<NVEC / 32, 256, 0, stream>>>(z, zbf);
  k_init_ema<<<(KCODES * DIM) / 256, 256, 0, stream>>>(ema_w, out_ema);
  k_vq_main <<<NVEC / 128, 256, 0, stream>>>(zbf, ebf, enorm,
                                             idx, out_idx, counts);
  k_scatter <<<(NVEC * DIM) / 256, 256, 0, stream>>>(z, emb, idx, out_main,
                                                     out_ema, &scal[0]);
  k_cs      <<<KCODES / 256, 256, 0, stream>>>(cluster_size, counts, cs, &scal[1]);
  k_final   <<<(KCODES * DIM) / 256, 256, 0, stream>>>(cs, scal, out_ema,
                                                       out_ncs, out_emb, out_loss);
}
